// AttnDecoderRNN_3178275799686
// MI455X (gfx1250) — compile-verified, matrix operand order checked by reading, not run on hardware
//
#include <hip/hip_runtime.h>
#include <hip/hip_bf16.h>
#include <math.h>

// Problem constants (from reference)
#define BB 32
#define SS 50
#define HH 256
#define VV 50257
#define TT 25
#define H2X (2 * HH)   // 512
#define H3X (3 * HH)   // 768

// ---------------- CDNA5 WMMA types ----------------
typedef __attribute__((ext_vector_type(16))) __bf16 v16bf;
typedef __attribute__((ext_vector_type(8)))  float  v8f;

union BfVec {
    uint4 q[2];   // two contiguous 16B chunks (8 bf16 each)
    v16bf v;      // 16 bf16 operand lanes for v_wmma_f32_16x16x32_bf16
};

// float -> bf16, round-to-nearest-even
__device__ __forceinline__ unsigned short f2bf(float f) {
    unsigned int u = __float_as_uint(f);
    unsigned int rb = ((u >> 16) & 1u) + 0x7FFFu;
    return (unsigned short)((u + rb) >> 16);
}

__device__ __forceinline__ float sigmoidf_(float x) {
    return 1.0f / (1.0f + expf(-x));
}

// ---------------- one-time: out_w fp32 -> bf16 ----------------
__global__ void cvt_outw_bf16(const float* __restrict__ w, unsigned short* __restrict__ wbf, int n) {
    int i = blockIdx.x * blockDim.x + threadIdx.x;
    if (i < n) wbf[i] = f2bf(w[i]);
}

// ---------------- one-time: ua_keys[b,s,g] = sum_h enc[b,s,h]*Ua_w[g,h] + Ua_b[g] ----------------
__global__ void ua_keys_kernel(const float* __restrict__ enc,
                               const float* __restrict__ Ua_w,
                               const float* __restrict__ Ua_b,
                               float* __restrict__ ua_keys) {
    __shared__ float ek[HH];
    int row = blockIdx.x;          // b*S + s
    int g = threadIdx.x;           // 0..255
    ek[g] = enc[row * HH + g];
    __syncthreads();
    float acc = Ua_b[g];
    const float* wr = Ua_w + (size_t)g * HH;
    #pragma unroll 8
    for (int k = 0; k < HH; ++k) acc += wr[k] * ek[k];
    ua_keys[(size_t)row * HH + g] = acc;
}

// ---------------- per-step: attention + GRU (one block per batch row) ----------------
__global__ void step_kernel(const float* __restrict__ enc,        // [B,S,H]
                            const int*   __restrict__ target,     // [B,T]
                            const float* __restrict__ emb_table,  // [V,H]
                            const float* __restrict__ Wa_w, const float* __restrict__ Wa_b,
                            const float* __restrict__ ua_keys,    // [B,S,H]
                            const float* __restrict__ Va_w, const float* __restrict__ Va_b,
                            const float* __restrict__ w_ih, const float* __restrict__ b_ih,
                            const float* __restrict__ w_hh, const float* __restrict__ b_hh,
                            float* __restrict__ h,                // [B,H] state (in/out)
                            unsigned short* __restrict__ h2bf,    // [B,H] bf16 out
                            float* __restrict__ attn_out,         // d_out attn region [B,T,S]
                            float* __restrict__ hT_out,           // d_out hT region or null
                            int t) {
    __shared__ float h_s[HH];
    __shared__ float x_s[H2X];     // [emb | ctx]
    __shared__ float q_s[HH];
    __shared__ float w_s[SS];

    const int b = blockIdx.x;
    const int i = threadIdx.x;     // 0..255

    h_s[i] = h[(size_t)b * HH + i];
    int tok = (t == 0) ? 0 : target[b * TT + (t - 1)];   // teacher forcing, SOS=0
    x_s[i] = emb_table[(size_t)tok * HH + i];
    __syncthreads();

    // wa_q[i] = Wa_b[i] + sum_k h[k] * Wa_w[i,k]
    {
        float q = Wa_b[i];
        const float* wr = Wa_w + (size_t)i * HH;
        #pragma unroll 8
        for (int k = 0; k < HH; ++k) q += wr[k] * h_s[k];
        q_s[i] = q;
    }
    __syncthreads();

    // scores[s] = Va . tanh(q + ua_keys[b,s,:]) + Va_b   (8 waves x strided s)
    const int wave = i >> 5, lane = i & 31;
    for (int s = wave; s < SS; s += 8) {
        float p = 0.0f;
        const float* uk = ua_keys + ((size_t)b * SS + s) * HH;
        #pragma unroll 4
        for (int hh = lane; hh < HH; hh += 32)
            p += Va_w[hh] * tanhf(q_s[hh] + uk[hh]);
        #pragma unroll
        for (int off = 16; off > 0; off >>= 1) p += __shfl_down(p, off, 32);
        if (lane == 0) w_s[s] = p + Va_b[0];
    }
    __syncthreads();

    // softmax over S=50 (tiny; single thread)
    if (i == 0) {
        float m = w_s[0];
        for (int s = 1; s < SS; ++s) m = fmaxf(m, w_s[s]);
        float sum = 0.0f;
        for (int s = 0; s < SS; ++s) { float e = expf(w_s[s] - m); w_s[s] = e; sum += e; }
        float inv = 1.0f / sum;
        for (int s = 0; s < SS; ++s) w_s[s] *= inv;
    }
    __syncthreads();

    // ctx[i] = sum_s w[s] * enc[b,s,i]
    {
        float c = 0.0f;
        const float* eb = enc + (size_t)b * SS * HH + i;
        #pragma unroll 5
        for (int s = 0; s < SS; ++s) c += w_s[s] * eb[(size_t)s * HH];
        x_s[HH + i] = c;
    }
    __syncthreads();

    // GRU (PyTorch gate order r,z,n); each thread owns output channel i of each gate
    float gr = b_ih[i], gz = b_ih[HH + i], gn = b_ih[2 * HH + i];
    {
        const float* wr = w_ih + (size_t)i * H2X;
        const float* wz = w_ih + (size_t)(HH + i) * H2X;
        const float* wn = w_ih + (size_t)(2 * HH + i) * H2X;
        #pragma unroll 8
        for (int k = 0; k < H2X; ++k) {
            float xv = x_s[k];
            gr += wr[k] * xv; gz += wz[k] * xv; gn += wn[k] * xv;
        }
    }
    float hr = b_hh[i], hz = b_hh[HH + i], hn = b_hh[2 * HH + i];
    {
        const float* wr = w_hh + (size_t)i * HH;
        const float* wz = w_hh + (size_t)(HH + i) * HH;
        const float* wn = w_hh + (size_t)(2 * HH + i) * HH;
        #pragma unroll 8
        for (int k = 0; k < HH; ++k) {
            float hv = h_s[k];
            hr += wr[k] * hv; hz += wz[k] * hv; hn += wn[k] * hv;
        }
    }
    float r = sigmoidf_(gr + hr);
    float z = sigmoidf_(gz + hz);
    float n = tanhf(gn + r * hn);
    float h2 = (1.0f - z) * n + z * h_s[i];

    h[(size_t)b * HH + i] = h2;
    h2bf[(size_t)b * HH + i] = f2bf(h2);
    if (i < SS) attn_out[((size_t)b * TT + t) * SS + i] = w_s[i];
    if (hT_out) hT_out[(size_t)b * HH + i] = h2;
}

// ---------------- per-step: logits = h2 @ out_w^T + out_b  via v_wmma_f32_16x16x32_bf16 ----------------
// Each wave owns one N=16 vocab tile, computes full M=32 with two accumulators sharing B.
// A layout (16x32 bf16): lane L -> M=L&15, K chunks [koff..koff+7],[koff+16..koff+23], koff=(L>>4)*8.
// B layout (32x16 bf16): lane L -> N=L&15, same K chunking.  C/D: VGPR r -> M=r+8*(L>>4), N=L&15.
__global__ void logits_wmma(const unsigned short* __restrict__ h2bf,  // [32,256] bf16
                            const unsigned short* __restrict__ wbf,   // [V,256] bf16
                            const float* __restrict__ out_b,          // [V]
                            float* __restrict__ dec_out,              // [B,T,V]
                            int t) {
    const int wave = threadIdx.x >> 5;
    const int lane = threadIdx.x & 31;
    const int ntile = blockIdx.x * 8 + wave;
    const int nbase = ntile * 16;
    if (nbase >= VV) return;                      // wave-uniform exit (EXEC stays all-ones)

    const int ncol = lane & 15;
    const int nglob = nbase + ncol;
    const int ncl = (nglob < VV) ? nglob : (VV - 1);   // clamp loads on ragged last tile
    const int koff = (lane >> 4) * 8;
    const int m0 = lane & 15;

    v8f acc0 = {}; // M = 0..15
    v8f acc1 = {}; // M = 16..31

    const unsigned short* arow0 = h2bf + (size_t)m0 * HH;
    const unsigned short* arow1 = h2bf + (size_t)(m0 + 16) * HH;
    const unsigned short* brow  = wbf  + (size_t)ncl * HH;

    #pragma unroll
    for (int kb = 0; kb < HH; kb += 32) {
        BfVec a0, a1, bm;
        a0.q[0] = *(const uint4*)(arow0 + kb + koff);
        a0.q[1] = *(const uint4*)(arow0 + kb + koff + 16);
        a1.q[0] = *(const uint4*)(arow1 + kb + koff);
        a1.q[1] = *(const uint4*)(arow1 + kb + koff + 16);
        bm.q[0] = *(const uint4*)(brow + kb + koff);
        bm.q[1] = *(const uint4*)(brow + kb + koff + 16);
        acc0 = __builtin_amdgcn_wmma_f32_16x16x32_bf16(false, a0.v, false, bm.v,
                                                       (short)0, acc0, false, false);
        acc1 = __builtin_amdgcn_wmma_f32_16x16x32_bf16(false, a1.v, false, bm.v,
                                                       (short)0, acc1, false, false);
    }

    if (nglob < VV) {
        const float bias = out_b[nglob];
        const int mhi = (lane >> 4) * 8;
        #pragma unroll
        for (int r = 0; r < 8; ++r) {
            int m = r + mhi;                       // batch row for acc0
            dec_out[((size_t)m * TT + t) * VV + nglob] = acc0[r] + bias;
            dec_out[((size_t)(m + 16) * TT + t) * VV + nglob] = acc1[r] + bias;
        }
    }
}

// ---------------- per-step: in-place log_softmax over V (one block per batch row) ----------------
__global__ void logsoftmax_kernel(float* __restrict__ dec_out, int t) {
    __shared__ float sm[256], ssum[256];
    const int b = blockIdx.x;
    const int i = threadIdx.x;
    float* row = dec_out + ((size_t)b * TT + t) * VV;

    // online max / sum-exp
    float m = -INFINITY, s = 0.0f;
    for (int v = i; v < VV; v += 256) {
        float x = row[v];
        if (x > m) { s *= expf(m - x); m = x; }
        s += expf(x - m);
    }
    sm[i] = m; ssum[i] = s;
    __syncthreads();
    for (int off = 128; off > 0; off >>= 1) {
        if (i < off) {
            float m2 = sm[i + off], s2 = ssum[i + off];
            float M = fmaxf(sm[i], m2);
            ssum[i] = ssum[i] * expf(sm[i] - M) + s2 * expf(m2 - M);
            sm[i] = M;
        }
        __syncthreads();
    }
    const float lse = sm[0] + logf(ssum[0]);
    for (int v = i; v < VV; v += 256) row[v] -= lse;
}

// ---------------- launcher ----------------
extern "C" void kernel_launch(void* const* d_in, const int* in_sizes, int n_in,
                              void* d_out, int out_size, void* d_ws, size_t ws_size,
                              hipStream_t stream) {
    (void)in_sizes; (void)n_in; (void)out_size; (void)ws_size;

    const float* enc_hidden = (const float*)d_in[0];   // [1,B,H]
    const float* enc_out    = (const float*)d_in[1];   // [B,S,H]
    const int*   target     = (const int*)  d_in[2];   // [B,T]
    const float* embedding  = (const float*)d_in[3];   // [V,H]
    const float* Wa_w = (const float*)d_in[4];
    const float* Wa_b = (const float*)d_in[5];
    const float* Ua_w = (const float*)d_in[6];
    const float* Ua_b = (const float*)d_in[7];
    const float* Va_w = (const float*)d_in[8];
    const float* Va_b = (const float*)d_in[9];
    const float* w_ih = (const float*)d_in[10];
    const float* b_ih = (const float*)d_in[11];
    const float* w_hh = (const float*)d_in[12];
    const float* b_hh = (const float*)d_in[13];
    const float* out_w = (const float*)d_in[14];
    const float* out_b = (const float*)d_in[15];

    // d_out layout: dec_out [B,T,V] | hT [1,B,H] | attentions [B,T,S]
    float* dec_out  = (float*)d_out;
    float* hT_out   = dec_out + (size_t)BB * TT * VV;
    float* attn_out = hT_out + (size_t)BB * HH;

    // workspace layout (all 256B aligned)
    char* ws = (char*)d_ws;
    unsigned short* wbf    = (unsigned short*)ws;                       // V*H bf16 = 25,731,584 B
    float*          uakeys = (float*)(ws + (size_t)VV * HH * 2);        // B*S*H f32 = 1,638,400 B
    float*          hbuf   = (float*)((char*)uakeys + (size_t)BB * SS * HH * 4);   // B*H f32
    unsigned short* h2bf   = (unsigned short*)((char*)hbuf + (size_t)BB * HH * 4); // B*H bf16

    // init hidden state from encoder_hidden[0]
    hipMemcpyAsync(hbuf, enc_hidden, (size_t)BB * HH * sizeof(float),
                   hipMemcpyDeviceToDevice, stream);

    // one-time conversions / precompute
    {
        int n = VV * HH;
        cvt_outw_bf16<<<(n + 255) / 256, 256, 0, stream>>>(out_w, wbf, n);
        ua_keys_kernel<<<BB * SS, HH, 0, stream>>>(enc_out, Ua_w, Ua_b, uakeys);
    }

    const int ntiles = (VV + 15) / 16;          // 3142
    const int lblocks = (ntiles + 7) / 8;       // 393

    for (int t = 0; t < TT; ++t) {
        step_kernel<<<BB, HH, 0, stream>>>(enc_out, target, embedding,
                                           Wa_w, Wa_b, uakeys, Va_w, Va_b,
                                           w_ih, b_ih, w_hh, b_hh,
                                           hbuf, h2bf, attn_out,
                                           (t == TT - 1) ? hT_out : (float*)nullptr, t);
        logits_wmma<<<lblocks, 256, 0, stream>>>(h2bf, wbf, out_b, dec_out, t);
        logsoftmax_kernel<<<BB, 256, 0, stream>>>(dec_out, t);
    }
}
